// MultiHeadAttention_6700148982005
// MI455X (gfx1250) — matrix operand order repaired and checked by reading, not compiled, not run on hardware
//
#include <hip/hip_runtime.h>

// ---------------- types ----------------
typedef __attribute__((ext_vector_type(16))) __bf16 v16bf;
typedef __attribute__((ext_vector_type(8)))  __bf16 v8bf;
typedef __attribute__((ext_vector_type(8)))  float  v8f;

#define D_MODEL 1024
#define SEQ     2048
#define NHEAD   16
#define HDIM    64
#define NTOK    4096            // B*S = 2*2048

__device__ __forceinline__ unsigned short f2bf(float f) {
    unsigned int u = __float_as_uint(f);
    unsigned int r = u + 0x7FFFu + ((u >> 16) & 1u);   // round-to-nearest-even
    return (unsigned short)(r >> 16);
}

__device__ __forceinline__ v16bf mk16(v8bf lo, v8bf hi) {
    return __builtin_shufflevector(lo, hi, 0,1,2,3,4,5,6,7,8,9,10,11,12,13,14,15);
}

__device__ __forceinline__ v8bf ld8(const unsigned short* p) {
    return *reinterpret_cast<const v8bf*>(p);          // 16B-aligned by construction
}

__device__ __forceinline__ v8f wmma_bf16(v16bf a, v16bf b, v8f c) {
    return __builtin_amdgcn_wmma_f32_16x16x32_bf16(
        /*neg_a=*/false, a, /*neg_b=*/false, b,
        /*c_mod=*/(short)0, c, /*reuse_a=*/false, /*reuse_b=*/false);
}

// A fragment: 16 contiguous halves at lane half offset
__device__ __forceinline__ v16bf ldA(const unsigned short* p, int k) {
    return mk16(ld8(p + k), ld8(p + k + 8));
}
// B fragment: 16 contiguous halves
__device__ __forceinline__ v16bf ldB(const unsigned short* p, int k) {
    return mk16(ld8(p + k), ld8(p + k + 8));
}

// ---------------- conversion kernels ----------------
__global__ void cvt_bf16_kernel(const float* __restrict__ in,
                                unsigned short* __restrict__ out, int n) {
    int i = blockIdx.x * blockDim.x + threadIdx.x;
    if (i < n) out[i] = f2bf(in[i]);
}

// out[n][k] = bf16(in[k][n])  (1024x1024), LDS-tiled so both sides coalesce
__global__ __launch_bounds__(256)
void transpose_cvt_kernel(const float* __restrict__ in,
                          unsigned short* __restrict__ out) {
    __shared__ float tile[32][33];
    const int tx = threadIdx.x & 31;
    const int ty = threadIdx.x >> 5;               // 0..7
    const int k0 = blockIdx.y * 32;
    const int n0 = blockIdx.x * 32;
#pragma unroll
    for (int j = 0; j < 4; ++j)
        tile[ty + 8 * j][tx] = in[(size_t)(k0 + ty + 8 * j) * D_MODEL + n0 + tx];
    __syncthreads();
#pragma unroll
    for (int j = 0; j < 4; ++j)
        out[(size_t)(n0 + ty + 8 * j) * D_MODEL + k0 + tx] =
            f2bf(tile[tx][ty + 8 * j]);
}

// ---------------- GEMM: C[m][n] = A[m][:] . Wt[n][:] + bias[n] ----------------
// A: [4096,1024] bf16 row-major.  Bt: [1024,1024] bf16 (transposed weight).
// Each wave: 32x64 C tile, software-pipelined (double-buffered fragments).
// mode 0: store bf16 as Q/K layout [B,H,S,Dh]
// mode 1: store bf16 as Vt layout [B,H,Dh,S]
// mode 2: store fp32 row-major [4096,1024]
__global__ __launch_bounds__(256)
void gemm_bf16_wmma(const unsigned short* __restrict__ A,
                    const unsigned short* __restrict__ Bt,
                    const float* __restrict__ bias,
                    void* __restrict__ outp, int mode) {
    const int lane = threadIdx.x & 31;
    const int wave = threadIdx.x >> 5;
    const int hf   = lane >> 4;
    const int ln   = lane & 15;

    const int m0 = blockIdx.y * 256 + wave * 32;
    const int n0 = blockIdx.x * 64;

    const unsigned short* ap0 = A + (size_t)(m0 + ln) * D_MODEL + hf * 16;
    const unsigned short* ap1 = ap0 + (size_t)16 * D_MODEL;
    const unsigned short* bp0 = Bt + (size_t)(n0 + ln) * D_MODEL + hf * 16;
    const unsigned short* bp1 = bp0 + (size_t)16 * D_MODEL;
    const unsigned short* bp2 = bp0 + (size_t)32 * D_MODEL;
    const unsigned short* bp3 = bp0 + (size_t)48 * D_MODEL;

    v8f acc0[4], acc1[4];
#pragma unroll
    for (int t = 0; t < 4; ++t) {
        acc0[t] = (v8f){0,0,0,0,0,0,0,0};
        acc1[t] = (v8f){0,0,0,0,0,0,0,0};
    }

    // prologue: stage k0 = 0
    v16bf a0 = ldA(ap0, 0), a1 = ldA(ap1, 0);
    v16bf b0 = ldB(bp0, 0), b1 = ldB(bp1, 0);
    v16bf b2 = ldB(bp2, 0), b3 = ldB(bp3, 0);

    for (int k0 = 32; k0 < D_MODEL; k0 += 32) {
        // prefetch next stage while current WMMAs execute
        v16bf na0 = ldA(ap0, k0), na1 = ldA(ap1, k0);
        v16bf nb0 = ldB(bp0, k0), nb1 = ldB(bp1, k0);
        v16bf nb2 = ldB(bp2, k0), nb3 = ldB(bp3, k0);

        acc0[0] = wmma_bf16(a0, b0, acc0[0]);
        acc1[0] = wmma_bf16(a1, b0, acc1[0]);
        acc0[1] = wmma_bf16(a0, b1, acc0[1]);
        acc1[1] = wmma_bf16(a1, b1, acc1[1]);
        acc0[2] = wmma_bf16(a0, b2, acc0[2]);
        acc1[2] = wmma_bf16(a1, b2, acc1[2]);
        acc0[3] = wmma_bf16(a0, b3, acc0[3]);
        acc1[3] = wmma_bf16(a1, b3, acc1[3]);

        a0 = na0; a1 = na1;
        b0 = nb0; b1 = nb1; b2 = nb2; b3 = nb3;
    }
    // epilogue: last stage
    acc0[0] = wmma_bf16(a0, b0, acc0[0]);
    acc1[0] = wmma_bf16(a1, b0, acc1[0]);
    acc0[1] = wmma_bf16(a0, b1, acc0[1]);
    acc1[1] = wmma_bf16(a1, b1, acc1[1]);
    acc0[2] = wmma_bf16(a0, b2, acc0[2]);
    acc1[2] = wmma_bf16(a1, b2, acc1[2]);
    acc0[3] = wmma_bf16(a0, b3, acc0[3]);
    acc1[3] = wmma_bf16(a1, b3, acc1[3]);

#pragma unroll
    for (int t = 0; t < 4; ++t) {
        const int col = n0 + t * 16 + ln;
        const float bv = bias[col];
#pragma unroll
        for (int g = 0; g < 2; ++g) {
#pragma unroll
            for (int r = 0; r < 8; ++r) {
                const int row  = m0 + g * 16 + r + hf * 8;   // token index
                const float v  = (g ? acc1[t][r] : acc0[t][r]) + bv;
                if (mode == 2) {
                    ((float*)outp)[(size_t)row * D_MODEL + col] = v;
                } else {
                    const int b = row >> 11, s = row & 2047;
                    const int h = col >> 6,  d = col & 63;
                    size_t idx;
                    if (mode == 0)  // [B,H,S,Dh]
                        idx = (((size_t)(b * NHEAD + h) * SEQ) + s) * HDIM + d;
                    else            // [B,H,Dh,S]
                        idx = (((size_t)(b * NHEAD + h) * HDIM) + d) * SEQ + s;
                    ((unsigned short*)outp)[idx] = f2bf(v);
                }
            }
        }
    }
}

// ---------------- Flash attention over one (b,h), 16 queries per wave ----------------
// Qm/Km: [B,H,S,Dh] bf16;  Vt: [B,H,Dh,S] bf16;  Ao: [4096,1024] bf16 (token-major)
// K fragments are double-buffered across kv iterations; V fragments prefetch
// behind the softmax VALU. No block barriers: each wave is self-contained.
__global__ __launch_bounds__(256)
void flash_attn_wmma(const unsigned short* __restrict__ Qm,
                     const unsigned short* __restrict__ Km,
                     const unsigned short* __restrict__ Vt,
                     unsigned short* __restrict__ Ao) {
    const int lane = threadIdx.x & 31;
    const int wave = threadIdx.x >> 5;
    const int hf   = lane >> 4;
    const int ln   = lane & 15;

    const int bh = blockIdx.y;                 // b*16+h
    const int b  = bh >> 4, h = bh & 15;
    const int q0 = blockIdx.x * 128 + wave * 16;

    const unsigned short* Qb  = Qm + (size_t)bh * SEQ * HDIM;
    const unsigned short* Kb  = Km + (size_t)bh * SEQ * HDIM;
    const unsigned short* Vtb = Vt + (size_t)bh * HDIM * SEQ;

    // per-wave LDS tile for P re-striping: 16 rows x 32 cols, stride 40 halves
    __shared__ __align__(16) unsigned short pbuf[8][16 * 40];

    // resident Q fragments (d 0..31 and 32..63)
    v16bf qf[2];
#pragma unroll
    for (int j = 0; j < 2; ++j) {
        const unsigned short* qp = Qb + (size_t)(q0 + ln) * HDIM + j * 32 + hf * 16;
        qf[j] = mk16(ld8(qp), ld8(qp + 8));
    }

    float m_run[8], l_run[8];
    v8f oacc[4];
#pragma unroll
    for (int r = 0; r < 8; ++r) { m_run[r] = -3.0e38f; l_run[r] = 0.0f; }
#pragma unroll
    for (int t = 0; t < 4; ++t) oacc[t] = (v8f){0,0,0,0,0,0,0,0};

    const float scale = 0.125f;                // 1/sqrt(64)

    // prologue: K fragments for kv0 = 0.  kf[tile*2+j]: tile = kv rows
    // {kv..kv+15, kv+16..kv+31}, j = d-chunk {0..31, 32..63}
    v16bf kf[4];
#pragma unroll
    for (int tile = 0; tile < 2; ++tile)
#pragma unroll
        for (int j = 0; j < 2; ++j) {
            const unsigned short* kp =
                Kb + (size_t)(tile * 16 + ln) * HDIM + j * 32 + hf * 16;
            kf[tile * 2 + j] = mk16(ld8(kp), ld8(kp + 8));
        }

    for (int kv0 = 0; kv0 < SEQ; kv0 += 32) {
        // ---- scores from resident K fragments ----
        v8f s0 = (v8f){0,0,0,0,0,0,0,0};
        v8f s1 = (v8f){0,0,0,0,0,0,0,0};
        s0 = wmma_bf16(qf[0], kf[0], s0);
        s0 = wmma_bf16(qf[1], kf[1], s0);
        s1 = wmma_bf16(qf[0], kf[2], s1);
        s1 = wmma_bf16(qf[1], kf[3], s1);

        // ---- prefetch V fragments; they complete behind the softmax VALU ----
        v16bf vb[4];
#pragma unroll
        for (int t = 0; t < 4; ++t) {
            const unsigned short* vp =
                Vtb + (size_t)(t * 16 + ln) * SEQ + kv0 + hf * 16;
            vb[t] = mk16(ld8(vp), ld8(vp + 8));
        }

        // ---- prefetch next iteration's K fragments (wrap-guarded) ----
        const int kvn = (kv0 + 32 < SEQ) ? (kv0 + 32) : 0;
        v16bf nkf[4];
#pragma unroll
        for (int tile = 0; tile < 2; ++tile)
#pragma unroll
            for (int j = 0; j < 2; ++j) {
                const unsigned short* kp =
                    Kb + (size_t)(kvn + tile * 16 + ln) * HDIM + j * 32 + hf * 16;
                nkf[tile * 2 + j] = mk16(ld8(kp), ld8(kp + 8));
            }

        s0 *= scale;
        s1 *= scale;

        // ---- online softmax per row (rows live per-VGPR, cols across 16 lanes) ----
        float corr[8];
#pragma unroll
        for (int r = 0; r < 8; ++r) {
            float mx = fmaxf(s0[r], s1[r]);
            mx = fmaxf(mx, __shfl_xor(mx, 1));
            mx = fmaxf(mx, __shfl_xor(mx, 2));
            mx = fmaxf(mx, __shfl_xor(mx, 4));
            mx = fmaxf(mx, __shfl_xor(mx, 8));
            const float mnew = fmaxf(m_run[r], mx);
            const float c    = __expf(m_run[r] - mnew);
            const float p0   = __expf(s0[r] - mnew);
            const float p1   = __expf(s1[r] - mnew);
            float rs = p0 + p1;
            rs += __shfl_xor(rs, 1);
            rs += __shfl_xor(rs, 2);
            rs += __shfl_xor(rs, 4);
            rs += __shfl_xor(rs, 8);
            l_run[r] = l_run[r] * c + rs;
            m_run[r] = mnew;
            corr[r]  = c;
            const int row = r + hf * 8;
            pbuf[wave][row * 40 + ln]      = f2bf(p0);
            pbuf[wave][row * 40 + 16 + ln] = f2bf(p1);
        }
#pragma unroll
        for (int t = 0; t < 4; ++t)
#pragma unroll
            for (int r = 0; r < 8; ++r) oacc[t][r] *= corr[r];

        // pbuf region is per-wave: same-wave DS ordering + backend dscnt wait
        // suffice; this only pins the scheduler's program order.
        __builtin_amdgcn_wave_barrier();

        // ---- P fragment (16q x 32kv) from LDS, then PV WMMAs ----
        const unsigned short* pp = &pbuf[wave][ln * 40 + hf * 16];
        v16bf pa = mk16(ld8(pp), ld8(pp + 8));
#pragma unroll
        for (int t = 0; t < 4; ++t)
            oacc[t] = wmma_bf16(pa, vb[t], oacc[t]);

        __builtin_amdgcn_wave_barrier();

        kf[0] = nkf[0]; kf[1] = nkf[1]; kf[2] = nkf[2]; kf[3] = nkf[3];
    }

    // ---- normalize and store to token-major [4096,1024] bf16 ----
#pragma unroll
    for (int t = 0; t < 4; ++t)
#pragma unroll
        for (int r = 0; r < 8; ++r) {
            const int srow = q0 + r + hf * 8;            // sequence position
            const int col  = h * HDIM + t * 16 + ln;     // feature column
            const float val = oacc[t][r] / l_run[r];
            Ao[((size_t)(b * SEQ + srow)) * D_MODEL + col] = f2bf(val);
        }
}

// ---------------- launch ----------------
extern "C" void kernel_launch(void* const* d_in, const int* in_sizes, int n_in,
                              void* d_out, int out_size, void* d_ws, size_t ws_size,
                              hipStream_t stream) {
    const float* x  = (const float*)d_in[0];
    const float* Wq = (const float*)d_in[1];
    const float* bq = (const float*)d_in[2];
    const float* Wk = (const float*)d_in[3];
    const float* bk = (const float*)d_in[4];
    const float* Wv = (const float*)d_in[5];
    const float* bv = (const float*)d_in[6];
    const float* Wo = (const float*)d_in[7];
    const float* bo = (const float*)d_in[8];

    unsigned short* ws = (unsigned short*)d_ws;
    const size_t M1 = 1048576;                  // 1024*1024 elements
    unsigned short* xb   = ws;                  // 4M elems
    unsigned short* wqt  = ws + 4 * M1;
    unsigned short* wkt  = ws + 5 * M1;
    unsigned short* wvt  = ws + 6 * M1;
    unsigned short* wot  = ws + 7 * M1;
    unsigned short* qbuf = ws + 8 * M1;         // 4M
    unsigned short* kbuf = ws + 12 * M1;        // 4M
    unsigned short* vtbf = ws + 16 * M1;        // 4M
    unsigned short* aobf = ws + 20 * M1;        // 4M  (total 48 MB)

    const int nx = NTOK * D_MODEL;              // 4,194,304
    cvt_bf16_kernel<<<nx / 256, 256, 0, stream>>>(x, xb, nx);
    dim3 tg(32, 32);
    transpose_cvt_kernel<<<tg, 256, 0, stream>>>(Wq, wqt);
    transpose_cvt_kernel<<<tg, 256, 0, stream>>>(Wk, wkt);
    transpose_cvt_kernel<<<tg, 256, 0, stream>>>(Wv, wvt);
    transpose_cvt_kernel<<<tg, 256, 0, stream>>>(Wo, wot);

    dim3 gg(D_MODEL / 64, NTOK / 256);          // (16, 16)
    gemm_bf16_wmma<<<gg, 256, 0, stream>>>(xb, wqt, bq, (void*)qbuf, 0);
    gemm_bf16_wmma<<<gg, 256, 0, stream>>>(xb, wkt, bk, (void*)kbuf, 0);
    gemm_bf16_wmma<<<gg, 256, 0, stream>>>(xb, wvt, bv, (void*)vtbf, 1);

    dim3 ga(SEQ / 128, 2 * NHEAD);              // (16, 32)
    flash_attn_wmma<<<ga, 256, 0, stream>>>(qbuf, kbuf, vtbf, aobf);

    gemm_bf16_wmma<<<gg, 256, 0, stream>>>(aobf, wot, bo, d_out, 2);
}
